// WipModule_75746043232416
// MI455X (gfx1250) — matrix operand order repaired and checked by reading, not compiled, not run on hardware
//
#include <hip/hip_runtime.h>
#include <hip/hip_bf16.h>

typedef __attribute__((ext_vector_type(16))) _Float16 v16h;
typedef __attribute__((ext_vector_type(8)))  float    v8f;
typedef __attribute__((ext_vector_type(4)))  unsigned u32x4;
typedef __attribute__((ext_vector_type(8)))  unsigned u32x8;
typedef __attribute__((ext_vector_type(4)))  int      v4i;

#define AS1 __attribute__((address_space(1)))
#define AS3 __attribute__((address_space(3)))

#define BN_SCALE 0.9999950000374997f

#if defined(__has_builtin)
# if __has_builtin(__builtin_amdgcn_global_load_async_to_lds_b128)
#  define HAVE_ASYNC_LDS 1
# else
#  define HAVE_ASYNC_LDS 0
# endif
# if __has_builtin(__builtin_amdgcn_s_wait_asynccnt)
#  define HAVE_WAIT_ASYNC 1
# else
#  define HAVE_WAIT_ASYNC 0
# endif
#else
# define HAVE_ASYNC_LDS 0
# define HAVE_WAIT_ASYNC 0
#endif

__device__ __forceinline__ float lrelu(float x) { return x > 0.f ? x : 0.01f * x; }

__device__ __forceinline__ v8f wmma16(v16h a, v16h b, v8f c) {
    return __builtin_amdgcn_wmma_f32_16x16x32_f16(false, a, false, b, (short)0, c, false, false);
}

__device__ __forceinline__ void async_wait0() {
#if HAVE_WAIT_ASYNC
    __builtin_amdgcn_s_wait_asynccnt(0);
#else
    asm volatile("s_wait_asynccnt 0x0" ::: "memory");
#endif
}

// copy 16 bytes global -> LDS, async (per-lane GLOBAL_LOAD_ASYNC_TO_LDS_B128)
__device__ __forceinline__ void copy_b128(const _Float16* g, _Float16* l) {
#if HAVE_ASYNC_LDS
    __builtin_amdgcn_global_load_async_to_lds_b128((AS1 v4i*)g, (AS3 v4i*)l, 0, 0);
#else
    *(uint4*)l = *(const uint4*)g;
#endif
}

// A fragment (16x32 f16, M x K), LDS row-major [M][32].
// ISA 7.12.2: lanes 0-15 M=lane; VGPR0-3 hold K 0..7 (lanes<16) / 8..15 (lanes>=16),
// VGPR4-7 hold K 16..23 / 24..31.
__device__ __forceinline__ v16h frag_a(const _Float16* __restrict__ A, int m_base, int lane) {
    v16h f;
    const _Float16* p = A + (m_base + (lane & 15)) * 32;
    const int g8 = (lane >> 4) * 8;
#pragma unroll
    for (int v = 0; v < 8; ++v)  f[v] = p[v + g8];
#pragma unroll
    for (int v = 8; v < 16; ++v) f[v] = p[v + 8 + g8];
    return f;
}

// B fragment (32x16 f16, K x N), LDS column-major [N][32]:
// lanes 0-15 hold K=0..7,16..23 of column N=lane, lanes 16-31 hold K=8..15,24..31.
__device__ __forceinline__ v16h frag_b(const _Float16* __restrict__ B, int n_base, int lane) {
    v16h f;
    const _Float16* p = B + (n_base + (lane & 15)) * 32 + (lane >> 4) * 8;
#pragma unroll
    for (int v = 0; v < 16; ++v) f[v] = p[v < 8 ? v : v + 8];
    return f;
}

// ---------------- pack / convert kernels ----------------

__global__ void k_f32_to_f16(const float* __restrict__ src, _Float16* __restrict__ dst, int n) {
    for (int i = blockIdx.x * blockDim.x + threadIdx.x; i < n; i += gridDim.x * blockDim.x)
        dst[i] = (_Float16)src[i];
}

// fmap2 (128,32,48) -> f16 padded by 5 each side: (128,42,58)
__global__ void k_pad_fmap2(const float* __restrict__ f2, _Float16* __restrict__ dst) {
    const int n = 128 * 42 * 58;
    for (int i = blockIdx.x * blockDim.x + threadIdx.x; i < n; i += gridDim.x * blockDim.x) {
        int c = i / (42 * 58), r = i % (42 * 58);
        int ph = r / 58, pw = r % 58;
        int h = ph - 5, w = pw - 5;
        float v = ((unsigned)h < 32u && (unsigned)w < 48u) ? f2[(c * 32 + h) * 48 + w] : 0.f;
        dst[i] = (_Float16)v;
    }
}

// wd (Cin=64, Cout=32, 4, 4) -> [Cout][Cin*16] f16
__global__ void k_pack_wd(const float* __restrict__ wd, _Float16* __restrict__ dst) {
    const int n = 32 * 1024;
    for (int i = blockIdx.x * blockDim.x + threadIdx.x; i < n; i += gridDim.x * blockDim.x) {
        int co = i >> 10, rest = i & 1023;           // rest = cin*16 + tap
        dst[i] = (_Float16)wd[((rest >> 4) * 32 + co) * 16 + (rest & 15)];
    }
}

// ---------------- conv1: fused mvol build + implicit GEMM + TDM staging --------
// M=96, K=2304 (=256ch * 3x3), N=1536 per displacement; 121 displacements.
__global__ __launch_bounds__(256)
void k_conv1(const _Float16* __restrict__ x1, const _Float16* __restrict__ x2p,
             const _Float16* __restrict__ w1, _Float16* __restrict__ y1) {
    __shared__ __align__(16) _Float16 sA[96 * 32];
    __shared__ __align__(16) _Float16 sB[128 * 32];
    const int t = threadIdx.x, lane = t & 31, wave = t >> 5;
    const int nb = blockIdx.x * 128;
    const int d = blockIdx.y;
    const int pi = d / 11, pj = d % 11;    // i: horizontal shift idx, j: vertical
    const int di = pi - 5, dj = pj - 5;

    const v8f zero = {0.f, 0.f, 0.f, 0.f, 0.f, 0.f, 0.f, 0.f};
    v8f acc[6];
#pragma unroll
    for (int mt = 0; mt < 6; ++mt) acc[mt] = zero;

    const int col = t >> 1;
    const int n = nb + col;
    const int h = n / 48, w = n % 48;
    const int khalf = (t & 1) * 16;
    const unsigned lds_a = (unsigned)(size_t)(AS3 _Float16*)sA;   // LDS byte offset of sA

    for (int kk = 0; kk < 2304; kk += 32) {
        // stage A via Tensor Data Mover: 2D tile, 96 rows x 32 f16, row stride 2304 f16.
        if (wave == 0) {
            unsigned long long ga = (unsigned long long)(size_t)(w1 + kk);
            u32x4 g0;
            g0[0] = 1u;                                            // count=1, user mode
            g0[1] = lds_a;                                         // lds_addr (bytes)
            g0[2] = (unsigned)ga;                                  // global_addr[31:0]
            g0[3] = (unsigned)((ga >> 32) & 0x01FFFFFFu) | 0x80000000u; // addr[56:32] | type=2
            u32x8 g1;
            g1[0] = 0x00010000u;                                   // data_size=1 (2 bytes)
            g1[1] = (2304u & 0xFFFFu) << 16;                       // tensor_dim0[15:0]
            g1[2] = (2304u >> 16) | ((4096u & 0xFFFFu) << 16);     // dim0[31:16] | tensor_dim1 lo
            g1[3] = 0u | (32u << 16);                              // dim1 hi | tile_dim0=32
            g1[4] = 96u;                                           // tile_dim1=96, tile_dim2=0
            g1[5] = 2304u;                                         // tensor_dim0_stride[31:0]
            g1[6] = 0u;                                            // stride hi | dim1_stride lo
            g1[7] = 0u;
            asm volatile("tensor_load_to_lds %0, %1\n\t"
                         "s_wait_tensorcnt 0x0"
                         :: "s"(g0), "s"(g1) : "memory");
        }
        // stage B: im2col of mvol (built on the fly, never materialized)
#pragma unroll
        for (int u = 0; u < 16; ++u) {
            int kl = khalf + u, kg = kk + kl;
            int c = kg / 9, rs = kg - 9 * c;
            int r = rs / 3, s = rs - 3 * r;
            int ch = h + r - 1, cw = w + s - 1;
            _Float16 v = (_Float16)0.f;
            if ((unsigned)ch < 32u && (unsigned)cw < 48u) {
                if (c < 128) {
                    // fmap1 channel, masked by displacement validity
                    if ((unsigned)(ch + dj) < 32u && (unsigned)(cw + di) < 48u)
                        v = x1[c * 1536 + ch * 48 + cw];
                } else {
                    // shifted fmap2 channel (pre-padded -> mask implicit)
                    v = x2p[(c - 128) * 2436 + (ch + pj) * 58 + (cw + pi)];
                }
            }
            sB[col * 32 + kl] = v;
        }
        __syncthreads();
        v16h bf = frag_b(sB, wave * 16, lane);
#pragma unroll
        for (int mt = 0; mt < 6; ++mt) {
            v16h af = frag_a(sA, mt * 16, lane);
            acc[mt] = wmma16(af, bf, acc[mt]);
        }
        __syncthreads();
    }
    // epilogue: BN scale + leaky relu, f16 store
    const int g = lane >> 4, nl = lane & 15;
    _Float16* out = y1 + (size_t)d * 96 * 1536;
#pragma unroll
    for (int mt = 0; mt < 6; ++mt)
#pragma unroll
        for (int r8 = 0; r8 < 8; ++r8) {
            int co = mt * 16 + g * 8 + r8;
            out[co * 1536 + nb + wave * 16 + nl] = (_Float16)lrelu(acc[mt][r8] * BN_SCALE);
        }
}

// ---------------- generic 3x3 conv (layers 2-4), implicit GEMM + async staging --
template <int MT, int STRIDE>
__global__ __launch_bounds__(256)
void k_conv(const _Float16* __restrict__ in, const _Float16* __restrict__ wt,
            _Float16* __restrict__ out, int Ci, int Hin, int Win, int Hout, int Wout) {
    __shared__ __align__(16) _Float16 sA[128 * 32];
    __shared__ __align__(16) _Float16 sB[128 * 32];
    const int K = Ci * 9;
    const int N = Hout * Wout;
    const int t = threadIdx.x, lane = t & 31, wave = t >> 5;
    const int nb = blockIdx.x * 128;
    const int d = blockIdx.y;
    const _Float16* inD = in + (size_t)d * Ci * Hin * Win;

    const v8f zero = {0.f, 0.f, 0.f, 0.f, 0.f, 0.f, 0.f, 0.f};
    v8f acc[MT];
#pragma unroll
    for (int mt = 0; mt < MT; ++mt) acc[mt] = zero;

    const int col = t >> 1, n = nb + col;
    const int oh = n / Wout, ow = n % Wout;
    const int khalf = (t & 1) * 16;
    const int chunks = MT * 16 * 4;                   // Co rows * 4 x b128 per row

    for (int kk = 0; kk < K; kk += 32) {
        // stage A (weights) with GLOBAL_LOAD_ASYNC_TO_LDS_B128 (16B chunks)
        for (int idx = t; idx < chunks; idx += 256)
            copy_b128(wt + (idx >> 2) * K + kk + (idx & 3) * 8, &sA[idx * 8]);
        // stage B: im2col gather
#pragma unroll
        for (int u = 0; u < 16; ++u) {
            int kl = khalf + u, kg = kk + kl;
            int c = kg / 9, rs = kg - 9 * c;
            int r = rs / 3, s = rs - 3 * r;
            int ih = oh * STRIDE + r - 1, iw = ow * STRIDE + s - 1;
            _Float16 v = (_Float16)0.f;
            if ((unsigned)ih < (unsigned)Hin && (unsigned)iw < (unsigned)Win)
                v = inD[(c * Hin + ih) * Win + iw];
            sB[col * 32 + kl] = v;
        }
#if HAVE_ASYNC_LDS
        async_wait0();
#endif
        __syncthreads();
        v16h bf = frag_b(sB, wave * 16, lane);
#pragma unroll
        for (int mt = 0; mt < MT; ++mt) {
            v16h af = frag_a(sA, mt * 16, lane);
            acc[mt] = wmma16(af, bf, acc[mt]);
        }
        __syncthreads();
    }
    const int g = lane >> 4, nl = lane & 15;
    _Float16* outD = out + (size_t)d * (MT * 16) * N;
#pragma unroll
    for (int mt = 0; mt < MT; ++mt)
#pragma unroll
        for (int r8 = 0; r8 < 8; ++r8) {
            int co = mt * 16 + g * 8 + r8;
            outD[co * N + nb + wave * 16 + nl] = (_Float16)lrelu(acc[mt][r8] * BN_SCALE);
        }
}

// ---------------- deconv (ConvTranspose2d 64->32, k4 s2 p1) as implicit GEMM ----
// out[oh,ow] = sum_{c,kh,kw} in[c, (oh+1-kh)/2, (ow+1-kw)/2] * w[c, co, kh, kw]
// M=32, K=64*16=1024, N=1536
__global__ __launch_bounds__(256)
void k_deconv(const _Float16* __restrict__ in, const _Float16* __restrict__ wt,
              _Float16* __restrict__ out) {
    __shared__ __align__(16) _Float16 sA[32 * 32];
    __shared__ __align__(16) _Float16 sB[128 * 32];
    const int t = threadIdx.x, lane = t & 31, wave = t >> 5;
    const int nb = blockIdx.x * 128;
    const int d = blockIdx.y;
    const _Float16* inD = in + (size_t)d * 64 * 384;

    const v8f zero = {0.f, 0.f, 0.f, 0.f, 0.f, 0.f, 0.f, 0.f};
    v8f acc[2] = {zero, zero};

    const int col = t >> 1, n = nb + col;
    const int oh = n / 48, ow = n % 48;
    const int khalf = (t & 1) * 16;

    for (int kk = 0; kk < 1024; kk += 32) {
        if (t < 128)                                   // 32 rows * 4 x b128
            copy_b128(wt + (t >> 2) * 1024 + kk + (t & 3) * 8, &sA[t * 8]);
#pragma unroll
        for (int u = 0; u < 16; ++u) {
            int kl = khalf + u, kg = kk + kl;
            int c = kg >> 4, tap = kg & 15;
            int kh = tap >> 2, kw = tap & 3;
            int ph = oh + 1 - kh, pw = ow + 1 - kw;
            _Float16 v = (_Float16)0.f;
            if (ph >= 0 && pw >= 0 && !(ph & 1) && !(pw & 1)) {
                int ih = ph >> 1, iw = pw >> 1;
                if (ih < 16 && iw < 24) v = inD[(c * 16 + ih) * 24 + iw];
            }
            sB[col * 32 + kl] = v;
        }
#if HAVE_ASYNC_LDS
        async_wait0();
#endif
        __syncthreads();
        v16h bf = frag_b(sB, wave * 16, lane);
        acc[0] = wmma16(frag_a(sA, 0, lane), bf, acc[0]);
        acc[1] = wmma16(frag_a(sA, 16, lane), bf, acc[1]);
        __syncthreads();
    }
    const int g = lane >> 4, nl = lane & 15;
    _Float16* outD = out + (size_t)d * 32 * 1536;
#pragma unroll
    for (int mt = 0; mt < 2; ++mt)
#pragma unroll
        for (int r8 = 0; r8 < 8; ++r8) {
            int co = mt * 16 + g * 8 + r8;
            outD[co * 1536 + nb + wave * 16 + nl] = (_Float16)lrelu(acc[mt][r8] * BN_SCALE);
        }
}

// ---------------- conv5 (32 -> 1 channel): plain VALU, fp32 out -------------
__global__ void k_conv5(const _Float16* __restrict__ y5, const float* __restrict__ w5,
                        const float* __restrict__ b5, float* __restrict__ outp) {
    int i = blockIdx.x * blockDim.x + threadIdx.x;
    if (i >= 121 * 1536) return;
    int d = i / 1536, n = i % 1536, h = n / 48, w = n % 48;
    const _Float16* inD = y5 + (size_t)d * 32 * 1536;
    float s = b5[0];
#pragma unroll
    for (int c = 0; c < 32; ++c)
#pragma unroll
        for (int r = 0; r < 3; ++r) {
            int ih = h + r - 1;
            if ((unsigned)ih >= 32u) continue;
#pragma unroll
            for (int ss = 0; ss < 3; ++ss) {
                int iw = w + ss - 1;
                if ((unsigned)iw >= 48u) continue;
                s += (float)inD[(c * 32 + ih) * 48 + iw] * w5[(c * 3 + r) * 3 + ss];
            }
        }
    outp[i] = s;
}

// ---------------- host launch ----------------

extern "C" void kernel_launch(void* const* d_in, const int* in_sizes, int n_in,
                              void* d_out, int out_size, void* d_ws, size_t ws_size,
                              hipStream_t stream) {
    const float* fmap1 = (const float*)d_in[0];
    const float* fmap2 = (const float*)d_in[1];
    const float* w1 = (const float*)d_in[2];
    const float* w2 = (const float*)d_in[3];
    const float* w3 = (const float*)d_in[4];
    const float* w4 = (const float*)d_in[5];
    const float* wd = (const float*)d_in[6];
    const float* w5 = (const float*)d_in[7];
    const float* b5 = (const float*)d_in[8];

    char* ws = (char*)d_ws;
    size_t off = 0;
    auto alloc = [&](size_t halves) {
        _Float16* p = (_Float16*)(ws + off);
        off += ((halves * 2 + 255) & ~(size_t)255);
        return p;
    };
    _Float16* x1h = alloc(128 * 32 * 48);          // f16 fmap1
    _Float16* x2p = alloc(128 * 42 * 58);          // f16 fmap2, pre-padded by RU/RV
    _Float16* w1h = alloc(96 * 2304);
    _Float16* w2h = alloc(128 * 864);
    _Float16* w3h = alloc(128 * 1152);
    _Float16* w4h = alloc(64 * 1152);
    _Float16* wdh = alloc(32 * 1024);
    _Float16* y1  = alloc((size_t)121 * 96 * 1536);
    _Float16* y2  = alloc((size_t)121 * 128 * 384);
    _Float16* y3  = alloc((size_t)121 * 128 * 384);
    _Float16* y4  = alloc((size_t)121 * 64 * 384);
    _Float16* y5  = alloc((size_t)121 * 32 * 1536);

    k_f32_to_f16<<<256, 256, 0, stream>>>(fmap1, x1h, 128 * 32 * 48);
    k_pad_fmap2<<<256, 256, 0, stream>>>(fmap2, x2p);
    k_f32_to_f16<<<256, 256, 0, stream>>>(w1, w1h, 96 * 2304);
    k_f32_to_f16<<<128, 256, 0, stream>>>(w2, w2h, 128 * 864);
    k_f32_to_f16<<<128, 256, 0, stream>>>(w3, w3h, 128 * 1152);
    k_f32_to_f16<<<128, 256, 0, stream>>>(w4, w4h, 64 * 1152);
    k_pack_wd<<<64, 256, 0, stream>>>(wd, wdh);

    // conv1: fused mvol + GEMM (TDM weight staging), grid = 12 N-tiles x 121 disp
    k_conv1<<<dim3(12, 121), 256, 0, stream>>>(x1h, x2p, w1h, y1);
    // conv2: 96 -> 128, stride 2, 32x48 -> 16x24
    k_conv<8, 2><<<dim3(3, 121), 256, 0, stream>>>(y1, w2h, y2, 96, 32, 48, 16, 24);
    // conv3: 128 -> 128
    k_conv<8, 1><<<dim3(3, 121), 256, 0, stream>>>(y2, w3h, y3, 128, 16, 24, 16, 24);
    // conv4: 128 -> 64
    k_conv<4, 1><<<dim3(3, 121), 256, 0, stream>>>(y3, w4h, y4, 128, 16, 24, 16, 24);
    // deconv: 64 -> 32, 16x24 -> 32x48
    k_deconv<<<dim3(12, 121), 256, 0, stream>>>(y4, wdh, y5);
    // conv5: 32 -> 1, fp32 output
    k_conv5<<<(121 * 1536 + 255) / 256, 256, 0, stream>>>(y5, w5, b5, (float*)d_out);
}
